// MSA_2843268350467
// MI455X (gfx1250) — matrix operand order repaired and checked by reading, not compiled, not run on hardware
//
#include <hip/hip_runtime.h>

typedef __attribute__((ext_vector_type(16))) _Float16 v16h;
typedef __attribute__((ext_vector_type(8)))  _Float16 v8h;
typedef __attribute__((ext_vector_type(8)))  float    v8f;

#define DIM   768
#define P_SEQ 2048
#define BATCH 4
#define HEADS 12
#define HD    64
#define MTOT  (BATCH * P_SEQ)   /* 8192 rows */

// ---------------------------------------------------------------------------
// WMMA fragment helpers (CDNA5 gfx1250, wave32, V_WMMA_F32_16X16X32_F16)
// A 16x32 f16 layout: lane l holds row m=l&15; halves 0..7 = K = ko..ko+7,
// halves 8..15 = K = 16+ko..16+ko+7, ko = (l>=16) ? 8 : 0.
// B 32x16 f16 layout: lane l holds col n=l&15; halves 0..7 = K = kb..kb+7,
// halves 8..15 = K = 16+kb..16+kb+7, kb = (l>=16) ? 8 : 0.
// C/D f32 16x16: vgpr r, lane l -> (m = r + 8*(l>>4), n = l&15).
// ---------------------------------------------------------------------------
__device__ __forceinline__ v16h ldfragA(const _Float16* p, int lane) {
  const int ko = (lane & 16) ? 8 : 0;
  v8h lo = *(const v8h*)(p + ko);
  v8h hi = *(const v8h*)(p + ko + 16);
  v16h a;
#pragma unroll
  for (int i = 0; i < 8; ++i) { a[i] = lo[i]; a[i + 8] = hi[i]; }
  return a;
}

__device__ __forceinline__ v16h ldfragB(const _Float16* p, int lane) {
  const int kb = (lane & 16) ? 8 : 0;
  v8h lo = *(const v8h*)(p + kb);
  v8h hi = *(const v8h*)(p + kb + 16);
  v16h b;
#pragma unroll
  for (int i = 0; i < 8; ++i) { b[i] = lo[i]; b[i + 8] = hi[i]; }
  return b;
}

__device__ __forceinline__ v8f wmma_f16(v16h a, v16h b, v8f c) {
  // 8 args: (neg_a, A, neg_b, B, c_mod, C, reuse_a, reuse_b)
  return __builtin_amdgcn_wmma_f32_16x16x32_f16(false, a, false, b, (short)0, c,
                                                false, false);
}

// ---------------------------------------------------------------------------
// Kernel 1: transpose+convert weights f32 [K][N] -> f16 [N][K]
// ---------------------------------------------------------------------------
__global__ __launch_bounds__(256) void wt_kernel(
    const float* __restrict__ W0, const float* __restrict__ W1,
    const float* __restrict__ W2, const float* __restrict__ W3,
    _Float16* __restrict__ T0, _Float16* __restrict__ T1,
    _Float16* __restrict__ T2, _Float16* __restrict__ T3) {
  const float* W; _Float16* T;
  switch (blockIdx.y) {
    case 0:  W = W0; T = T0; break;
    case 1:  W = W1; T = T1; break;
    case 2:  W = W2; T = T2; break;
    default: W = W3; T = T3; break;
  }
  int flat = blockIdx.x * 256 + threadIdx.x;     // coalesced read along N
  int k = flat / DIM, n = flat % DIM;
  T[(size_t)n * DIM + k] = (_Float16)W[flat];
}

// ---------------------------------------------------------------------------
// Kernel 2: LayerNorm, f32 in -> f16 out, one block (256 thr) per row
// ---------------------------------------------------------------------------
__global__ __launch_bounds__(256) void ln_kernel(
    const float* __restrict__ x, const float* __restrict__ w,
    const float* __restrict__ b, _Float16* __restrict__ xn) {
  __shared__ float s1[256], s2[256];
  const int row = blockIdx.x;
  const int t = threadIdx.x;
  const float* xr = x + (size_t)row * DIM;
  float v0 = xr[t], v1 = xr[t + 256], v2 = xr[t + 512];
  s1[t] = v0 + v1 + v2;
  s2[t] = v0 * v0 + v1 * v1 + v2 * v2;
  __syncthreads();
#pragma unroll
  for (int off = 128; off > 0; off >>= 1) {
    if (t < off) { s1[t] += s1[t + off]; s2[t] += s2[t + off]; }
    __syncthreads();
  }
  const float mean = s1[0] * (1.0f / DIM);
  const float var  = s2[0] * (1.0f / DIM) - mean * mean;
  const float rstd = rsqrtf(var + 1e-5f);
  _Float16* o = xn + (size_t)row * DIM;
  o[t]       = (_Float16)((v0 - mean) * rstd * w[t]       + b[t]);
  o[t + 256] = (_Float16)((v1 - mean) * rstd * w[t + 256] + b[t + 256]);
  o[t + 512] = (_Float16)((v2 - mean) * rstd * w[t + 512] + b[t + 512]);
}

// ---------------------------------------------------------------------------
// Kernel 3: QKV GEMM. M=8192, N=768, K=768, grid.z selects Q/K/V.
// Block = 128 thr (4 waves); wave tile 64x32 (4x2 WMMA accums); block 64x128.
// Q,K stored [B,H,P,64]; V stored transposed [B,H,64,P] (+bias bv).
// ---------------------------------------------------------------------------
__global__ __launch_bounds__(128) void qkv_kernel(
    const _Float16* __restrict__ xn, const _Float16* __restrict__ WT3,
    const float* __restrict__ bv, _Float16* __restrict__ Qo,
    _Float16* __restrict__ Ko, _Float16* __restrict__ Vt) {
  const int z    = blockIdx.z;                       // 0=Q 1=K 2=V
  const _Float16* WT = WT3 + (size_t)z * DIM * DIM;  // [N][K]
  const int wave = threadIdx.x >> 5;
  const int lane = threadIdx.x & 31;
  const int m0 = blockIdx.x * 64;
  const int n0 = blockIdx.y * 128 + wave * 32;
  const int mrow = lane & 15;

  v8f acc[4][2];
#pragma unroll
  for (int mt = 0; mt < 4; ++mt)
#pragma unroll
    for (int nt = 0; nt < 2; ++nt)
#pragma unroll
      for (int i = 0; i < 8; ++i) acc[mt][nt][i] = 0.0f;

#pragma unroll 2
  for (int k = 0; k < DIM; k += 32) {
    v16h af[4], bf[2];
#pragma unroll
    for (int mt = 0; mt < 4; ++mt) {
      const _Float16* pa = xn + (size_t)(m0 + mt * 16 + mrow) * DIM + k;
      __builtin_prefetch((const void*)(pa + 64), 0, 3);
      af[mt] = ldfragA(pa, lane);
    }
#pragma unroll
    for (int nt = 0; nt < 2; ++nt) {
      const _Float16* pb = WT + (size_t)(n0 + nt * 16 + mrow) * DIM + k;
      __builtin_prefetch((const void*)(pb + 64), 0, 3);
      bf[nt] = ldfragB(pb, lane);
    }
#pragma unroll
    for (int mt = 0; mt < 4; ++mt)
#pragma unroll
      for (int nt = 0; nt < 2; ++nt)
        acc[mt][nt] = wmma_f16(af[mt], bf[nt], acc[mt][nt]);
  }

  // ---- store epilogue: z is block-uniform, keep the branch OUTSIDE the
  //      unrolled loops so codegen is three straight-line store loops ----
  const int mhi  = 8 * (lane >> 4);
  const int ncol = lane & 15;
  if (z == 0) {
#pragma unroll
    for (int mt = 0; mt < 4; ++mt)
#pragma unroll
      for (int nt = 0; nt < 2; ++nt)
#pragma unroll
        for (int r = 0; r < 8; ++r) {
          const int m = m0 + mt * 16 + r + mhi;
          const int n = n0 + nt * 16 + ncol;
          const int bb = m >> 11, p = m & 2047;
          const int h = n >> 6, d = n & 63;
          Qo[(((size_t)(bb * HEADS + h)) * P_SEQ + p) * HD + d] =
              (_Float16)acc[mt][nt][r];
        }
  } else if (z == 1) {
#pragma unroll
    for (int mt = 0; mt < 4; ++mt)
#pragma unroll
      for (int nt = 0; nt < 2; ++nt)
#pragma unroll
        for (int r = 0; r < 8; ++r) {
          const int m = m0 + mt * 16 + r + mhi;
          const int n = n0 + nt * 16 + ncol;
          const int bb = m >> 11, p = m & 2047;
          const int h = n >> 6, d = n & 63;
          Ko[(((size_t)(bb * HEADS + h)) * P_SEQ + p) * HD + d] =
              (_Float16)acc[mt][nt][r];
        }
  } else {
#pragma unroll
    for (int mt = 0; mt < 4; ++mt)
#pragma unroll
      for (int nt = 0; nt < 2; ++nt)
#pragma unroll
        for (int r = 0; r < 8; ++r) {
          const int m = m0 + mt * 16 + r + mhi;
          const int n = n0 + nt * 16 + ncol;
          const int bb = m >> 11, p = m & 2047;
          const int h = n >> 6, d = n & 63;
          const float val = acc[mt][nt][r] + bv[n];
          Vt[(((size_t)(bb * HEADS + h)) * HD + d) * P_SEQ + p] =
              (_Float16)val;
        }
  }
}

// ---------------------------------------------------------------------------
// Kernel 4: flash attention per (b,h). Block = 128 thr (4 waves).
// Each wave owns 16 q-rows; block owns 64. Iterate kv in tiles of 64.
// S = Q K^T (WMMA), diag mask, online softmax, O += P V (WMMA, V transposed).
// ---------------------------------------------------------------------------
__global__ __launch_bounds__(128) void attn_kernel(
    const _Float16* __restrict__ Q, const _Float16* __restrict__ K,
    const _Float16* __restrict__ Vt, _Float16* __restrict__ AO) {
  const int bh = blockIdx.x;
  const int b = bh / HEADS, h = bh % HEADS;
  const _Float16* Qh = Q  + (size_t)bh * P_SEQ * HD;
  const _Float16* Kh = K  + (size_t)bh * P_SEQ * HD;
  const _Float16* Vh = Vt + (size_t)bh * HD * P_SEQ;
  const int wave = threadIdx.x >> 5;
  const int lane = threadIdx.x & 31;
  const int q0 = blockIdx.y * 64 + wave * 16;     // this wave's q-row base
  const int ncol = lane & 15;
  const int mhi  = 8 * (lane >> 4);

  __shared__ _Float16 ldsP[4][16][72];            // per-wave P^(f16), padded

  // Q fragments, loaded once (K-dim = head_dim = 64 -> 2 frags)
  v16h qf[2];
  qf[0] = ldfragA(Qh + (size_t)(q0 + ncol) * HD + 0,  lane);
  qf[1] = ldfragA(Qh + (size_t)(q0 + ncol) * HD + 32, lane);

  const float scale = 0.125f;                     // 1/sqrt(64)
  float mrun[8], lrun[8];
  v8f o[4];
#pragma unroll
  for (int r = 0; r < 8; ++r) { mrun[r] = -__builtin_inff(); lrun[r] = 0.0f; }
#pragma unroll
  for (int t = 0; t < 4; ++t)
#pragma unroll
    for (int i = 0; i < 8; ++i) o[t][i] = 0.0f;

  for (int kv0 = 0; kv0 < P_SEQ; kv0 += 64) {
    // ---- S tile: 16 q-rows x 64 kv-cols ----
    v8f s[4];
#pragma unroll
    for (int nt = 0; nt < 4; ++nt) {
#pragma unroll
      for (int i = 0; i < 8; ++i) s[nt][i] = 0.0f;
      const _Float16* kp = Kh + (size_t)(kv0 + nt * 16 + ncol) * HD;
      s[nt] = wmma_f16(qf[0], ldfragB(kp + 0,  lane), s[nt]);
      s[nt] = wmma_f16(qf[1], ldfragB(kp + 32, lane), s[nt]);
    }

    // ---- scale + diagonal mask + per-row max ----
    float rmax[8];
#pragma unroll
    for (int r = 0; r < 8; ++r) rmax[r] = -__builtin_inff();
#pragma unroll
    for (int nt = 0; nt < 4; ++nt)
#pragma unroll
      for (int r = 0; r < 8; ++r) {
        float v = s[nt][r] * scale;
        const int m = q0 + r + mhi;
        const int n = kv0 + nt * 16 + ncol;
        if (m == n) v = -__builtin_inff();        // mask_self
        s[nt][r] = v;
        rmax[r] = fmaxf(rmax[r], v);
      }
#pragma unroll
    for (int r = 0; r < 8; ++r)
#pragma unroll
      for (int d = 1; d < 16; d <<= 1)
        rmax[r] = fmaxf(rmax[r], __shfl_xor(rmax[r], d));

    // ---- online softmax ----
    float scl[8], rsum[8];
#pragma unroll
    for (int r = 0; r < 8; ++r) {
      const float mn = fmaxf(mrun[r], rmax[r]);
      scl[r] = __expf(mrun[r] - mn);
      mrun[r] = mn;
      rsum[r] = 0.0f;
    }
#pragma unroll
    for (int nt = 0; nt < 4; ++nt)
#pragma unroll
      for (int r = 0; r < 8; ++r) {
        const float p = __expf(s[nt][r] - mrun[r]);
        s[nt][r] = p;
        rsum[r] += p;
      }
#pragma unroll
    for (int r = 0; r < 8; ++r)
#pragma unroll
      for (int d = 1; d < 16; d <<= 1)
        rsum[r] += __shfl_xor(rsum[r], d);
#pragma unroll
    for (int r = 0; r < 8; ++r) lrun[r] = lrun[r] * scl[r] + rsum[r];
#pragma unroll
    for (int t = 0; t < 4; ++t)
#pragma unroll
      for (int r = 0; r < 8; ++r) o[t][r] *= scl[r];

    // ---- C-layout -> A-layout reshape of P via LDS ----
    __syncthreads();                              // WAR vs previous frag reads
#pragma unroll
    for (int nt = 0; nt < 4; ++nt)
#pragma unroll
      for (int r = 0; r < 8; ++r)
        ldsP[wave][r + mhi][nt * 16 + ncol] = (_Float16)s[nt][r];
    __syncthreads();                              // RAW (cross-lane)

    v16h pf[2];
    pf[0] = ldfragA(&ldsP[wave][ncol][0],  lane);
    pf[1] = ldfragA(&ldsP[wave][ncol][32], lane);

    // ---- O += P x V (V transposed: [d][P]) ----
#pragma unroll
    for (int t = 0; t < 4; ++t) {
      const _Float16* vp = Vh + (size_t)(t * 16 + ncol) * P_SEQ + kv0;
      o[t] = wmma_f16(pf[0], ldfragB(vp + 0,  lane), o[t]);
      o[t] = wmma_f16(pf[1], ldfragB(vp + 32, lane), o[t]);
    }
  }

  // ---- normalize + store to [B*P, 768] f16 (h-major cols) ----
#pragma unroll
  for (int t = 0; t < 4; ++t)
#pragma unroll
    for (int r = 0; r < 8; ++r) {
      const float val = o[t][r] / lrun[r];
      const int p = q0 + r + mhi;
      const size_t row = (size_t)b * P_SEQ + p;
      const int col = h * HD + t * 16 + ncol;
      AO[row * DIM + col] = (_Float16)val;
    }
}

// ---------------------------------------------------------------------------
// Kernel 5: output projection. M=8192, N=768, K=768, out = AO @ Wfc + bfc (f32)
// ---------------------------------------------------------------------------
__global__ __launch_bounds__(128) void proj_kernel(
    const _Float16* __restrict__ A, const _Float16* __restrict__ WT,
    const float* __restrict__ bias, float* __restrict__ out) {
  const int wave = threadIdx.x >> 5;
  const int lane = threadIdx.x & 31;
  const int m0 = blockIdx.x * 64;
  const int n0 = blockIdx.y * 128 + wave * 32;
  const int mrow = lane & 15;

  v8f acc[4][2];
#pragma unroll
  for (int mt = 0; mt < 4; ++mt)
#pragma unroll
    for (int nt = 0; nt < 2; ++nt)
#pragma unroll
      for (int i = 0; i < 8; ++i) acc[mt][nt][i] = 0.0f;

#pragma unroll 2
  for (int k = 0; k < DIM; k += 32) {
    v16h af[4], bf[2];
#pragma unroll
    for (int mt = 0; mt < 4; ++mt) {
      const _Float16* pa = A + (size_t)(m0 + mt * 16 + mrow) * DIM + k;
      __builtin_prefetch((const void*)(pa + 64), 0, 3);
      af[mt] = ldfragA(pa, lane);
    }
#pragma unroll
    for (int nt = 0; nt < 2; ++nt) {
      const _Float16* pb = WT + (size_t)(n0 + nt * 16 + mrow) * DIM + k;
      __builtin_prefetch((const void*)(pb + 64), 0, 3);
      bf[nt] = ldfragB(pb, lane);
    }
#pragma unroll
    for (int mt = 0; mt < 4; ++mt)
#pragma unroll
      for (int nt = 0; nt < 2; ++nt)
        acc[mt][nt] = wmma_f16(af[mt], bf[nt], acc[mt][nt]);
  }

  const int mhi  = 8 * (lane >> 4);
  const int ncol = lane & 15;
#pragma unroll
  for (int mt = 0; mt < 4; ++mt)
#pragma unroll
    for (int nt = 0; nt < 2; ++nt)
#pragma unroll
      for (int r = 0; r < 8; ++r) {
        const int m = m0 + mt * 16 + r + mhi;
        const int n = n0 + nt * 16 + ncol;
        out[(size_t)m * DIM + n] = acc[mt][nt][r] + bias[n];
      }
}

// ---------------------------------------------------------------------------
extern "C" void kernel_launch(void* const* d_in, const int* in_sizes, int n_in,
                              void* d_out, int out_size, void* d_ws,
                              size_t ws_size, hipStream_t stream) {
  (void)in_sizes; (void)n_in; (void)out_size; (void)ws_size;
  const float* x   = (const float*)d_in[0];
  const float* lnw = (const float*)d_in[1];
  const float* lnb = (const float*)d_in[2];
  const float* Wq  = (const float*)d_in[3];
  const float* Wk  = (const float*)d_in[4];
  const float* Wv  = (const float*)d_in[5];
  const float* bv  = (const float*)d_in[6];
  const float* Wfc = (const float*)d_in[7];
  const float* bfc = (const float*)d_in[8];
  float* out = (float*)d_out;

  // workspace partition (all f16), total ~67.6 MB
  _Float16* ws = (_Float16*)d_ws;
  const size_t WSZ = (size_t)DIM * DIM;      // 589824 elems per weight
  const size_t ASZ = (size_t)MTOT * DIM;     // 6291456 elems per activation
  _Float16* WqkvT = ws;                      // 3 transposed weights, contiguous
  _Float16* WfcT  = WqkvT + 3 * WSZ;
  _Float16* xn    = WfcT + WSZ;
  _Float16* Qb    = xn + ASZ;
  _Float16* Kb    = Qb + ASZ;
  _Float16* Vt    = Kb + ASZ;
  _Float16* AO    = Vt + ASZ;

  wt_kernel<<<dim3(DIM * DIM / 256, 4), 256, 0, stream>>>(
      Wq, Wk, Wv, Wfc, WqkvT, WqkvT + WSZ, WqkvT + 2 * WSZ, WfcT);
  ln_kernel<<<MTOT, 256, 0, stream>>>(x, lnw, lnb, xn);
  qkv_kernel<<<dim3(MTOT / 64, DIM / 128, 3), 128, 0, stream>>>(
      xn, WqkvT, bv, Qb, Kb, Vt);
  attn_kernel<<<dim3(BATCH * HEADS, P_SEQ / 64), 128, 0, stream>>>(
      Qb, Kb, Vt, AO);
  proj_kernel<<<dim3(MTOT / 64, DIM / 128), 128, 0, stream>>>(
      AO, WfcT, bfc, out);
}
